// DeepSeekAttention_87067577024919
// MI455X (gfx1250) — matrix operand order repaired and checked by reading, not compiled, not run on hardware
//
#include <hip/hip_runtime.h>
#include <hip/hip_bf16.h>

typedef _Float16 f16;
typedef __attribute__((ext_vector_type(16))) _Float16 v16h;
typedef __attribute__((ext_vector_type(8)))  float    v8f;

#define EN   2048
#define HN   16
#define DN   128
#define SN   2048
#define BN   2
#define TOK  (BN * SN)        // 4096
#define QRN  64
#define KRN  128
#define KVW  (KRN + HN * 2 * DN)  // 4224
#define RDN  64

static __device__ __forceinline__ v8f wmma_f16(const v16h& a, const v16h& b, const v8f& c) {
    return __builtin_amdgcn_wmma_f32_16x16x32_f16(false, a, false, b, (short)0, c, false, false);
}

// async copy of 16 bytes (per lane) global -> LDS, tracked by ASYNCcnt
static __device__ __forceinline__ void async_b128(unsigned lds_off, const void* gaddr) {
    asm volatile("global_load_async_to_lds_b128 %0, %1, off"
                 :: "v"(lds_off), "v"((unsigned long long)(uintptr_t)gaddr)
                 : "memory");
}
static __device__ __forceinline__ void wait_async0() {
    asm volatile("s_wait_asynccnt 0x0" ::: "memory");
}
static __device__ __forceinline__ unsigned lds_off_u32(const void* p) {
    return (unsigned)(uintptr_t)p;   // generic LDS ptr: low 32 bits = LDS byte offset
}

// ---------------------------------------------------------------------------
// fp32 -> f16 convert
// ---------------------------------------------------------------------------
__global__ void cvt_f32_f16(const float* __restrict__ s, f16* __restrict__ d, size_t n) {
    size_t i = (size_t)blockIdx.x * blockDim.x + threadIdx.x;
    if (i < n) d[i] = (f16)s[i];
}

// ---------------------------------------------------------------------------
// Fused LayerNorm for both gains (mean/var shared), f16 outputs
// ---------------------------------------------------------------------------
__global__ __launch_bounds__(256) void ln_kernel(const float* __restrict__ x,
                                                 const float* __restrict__ gq,
                                                 const float* __restrict__ gkv,
                                                 f16* __restrict__ oq, f16* __restrict__ okv) {
    const int t = blockIdx.x;
    const float* row = x + (size_t)t * EN;
    float s = 0.f, s2 = 0.f;
    for (int j = threadIdx.x; j < EN; j += 256) {
        float v = row[j];
        s += v; s2 += v * v;
    }
    __shared__ float red0[256];
    __shared__ float red1[256];
    red0[threadIdx.x] = s; red1[threadIdx.x] = s2;
    __syncthreads();
    for (int off = 128; off > 0; off >>= 1) {
        if (threadIdx.x < off) {
            red0[threadIdx.x] += red0[threadIdx.x + off];
            red1[threadIdx.x] += red1[threadIdx.x + off];
        }
        __syncthreads();
    }
    const float mu  = red0[0] * (1.0f / EN);
    const float var = red1[0] * (1.0f / EN) - mu * mu;
    const float inv = rsqrtf(var + 1e-5f);
    for (int j = threadIdx.x; j < EN; j += 256) {
        float v = (row[j] - mu) * inv;
        oq [(size_t)t * EN + j] = (f16)(v * gq[j]);
        okv[(size_t)t * EN + j] = (f16)(v * gkv[j]);
    }
}

// ---------------------------------------------------------------------------
// WMMA GEMM, 64x64 block tile (for N==64 case). 128 threads / 4 waves.
// ---------------------------------------------------------------------------
template <typename OutT>
__global__ __launch_bounds__(128) void gemm64_f16(const f16* __restrict__ A, int lda,
                                                  const f16* __restrict__ B, int ldb,
                                                  OutT* __restrict__ C, int ldc,
                                                  int M, int N, int K) {
    __shared__ f16 lds_a[64][32];
    __shared__ f16 lds_bt[64][32];

    const int tid  = threadIdx.x;
    const int wave = tid >> 5, lane = tid & 31;
    const int lh   = lane & 15, hi = lane >> 4;
    const int row0 = blockIdx.y * 64, col0 = blockIdx.x * 64;
    const int wr   = (wave >> 1) * 32, wc = (wave & 1) * 32;

    v8f acc[2][2];
    { v8f z = {}; acc[0][0] = z; acc[0][1] = z; acc[1][0] = z; acc[1][1] = z; }

    for (int k0 = 0; k0 < K; k0 += 32) {
        {   // A tile 64x32 via async global->LDS (2 x b128 per thread)
            const int r  = tid >> 1;
            const int cs = (tid & 1) * 16;
            const f16* src = A + (size_t)(row0 + r) * lda + k0 + cs;
            unsigned la = lds_off_u32(&lds_a[r][cs]);
            async_b128(la,       src);
            async_b128(la + 16u, src + 8);
        }
        {   // B tile 32x64, transposed into LDS via VGPRs
            const int r2 = tid >> 2;
            const int c2 = (tid & 3) * 16;
            const f16* bsrc = B + (size_t)(k0 + r2) * ldb + col0 + c2;
            f16 tmp[16];
            ((int4*)tmp)[0] = ((const int4*)bsrc)[0];
            ((int4*)tmp)[1] = ((const int4*)bsrc)[1];
            if (k0 + 32 < K) __builtin_prefetch(bsrc + 32 * ldb, 0, 0);
#pragma unroll
            for (int i = 0; i < 16; ++i) lds_bt[c2 + i][r2] = tmp[i];
        }
        wait_async0();
        __syncthreads();

        v16h af[2], bf[2];
#pragma unroll
        for (int ti = 0; ti < 2; ++ti) {
            const f16* p = &lds_a[wr + ti * 16 + lh][hi ? 8 : 0];
#pragma unroll
            for (int i = 0; i < 8; ++i) { af[ti][i] = p[i]; af[ti][i + 8] = p[i + 16]; }
        }
#pragma unroll
        for (int tj = 0; tj < 2; ++tj) {
            const f16* p = &lds_bt[wc + tj * 16 + lh][hi ? 8 : 0];
#pragma unroll
            for (int i = 0; i < 8; ++i) { bf[tj][i] = p[i]; bf[tj][i + 8] = p[i + 16]; }
        }
#pragma unroll
        for (int ti = 0; ti < 2; ++ti)
#pragma unroll
            for (int tj = 0; tj < 2; ++tj)
                acc[ti][tj] = wmma_f16(af[ti], bf[tj], acc[ti][tj]);
        __syncthreads();
    }

#pragma unroll
    for (int ti = 0; ti < 2; ++ti)
#pragma unroll
        for (int tj = 0; tj < 2; ++tj)
#pragma unroll
            for (int r = 0; r < 8; ++r) {
                const int gm = row0 + wr + ti * 16 + (hi ? r + 8 : r);
                const int gn = col0 + wc + tj * 16 + lh;
                C[(size_t)gm * ldc + gn] = (OutT)acc[ti][tj][r];
            }
}

// ---------------------------------------------------------------------------
// WMMA GEMM, 128x128 block tile. 256 threads / 8 waves, each wave 32x64.
// M,N multiples of 128; K multiple of 32.
// ---------------------------------------------------------------------------
template <typename OutT>
__global__ __launch_bounds__(256) void gemm128_f16(const f16* __restrict__ A, int lda,
                                                   const f16* __restrict__ B, int ldb,
                                                   OutT* __restrict__ C, int ldc,
                                                   int M, int N, int K) {
    __shared__ f16 lds_a[128][32];
    __shared__ f16 lds_bt[128][32];

    const int tid  = threadIdx.x;
    const int wave = tid >> 5, lane = tid & 31;
    const int lh   = lane & 15, hi = lane >> 4;
    const int row0 = blockIdx.y * 128, col0 = blockIdx.x * 128;
    const int wr   = (wave >> 1) * 32;     // 4 row-bands of 32
    const int wc   = (wave & 1) * 64;      // 2 col-bands of 64

    v8f acc[2][4];
    {
        v8f z = {};
#pragma unroll
        for (int i = 0; i < 2; ++i)
#pragma unroll
            for (int j = 0; j < 4; ++j) acc[i][j] = z;
    }

    for (int k0 = 0; k0 < K; k0 += 32) {
        {   // A tile 128x32 via async global->LDS (2 x b128 per thread)
            const int r  = tid >> 1;
            const int cs = (tid & 1) * 16;
            const f16* src = A + (size_t)(row0 + r) * lda + k0 + cs;
            unsigned la = lds_off_u32(&lds_a[r][cs]);
            async_b128(la,       src);
            async_b128(la + 16u, src + 8);
        }
        {   // B tile 32x128, transposed into LDS via VGPRs
            const int r2 = tid >> 3;            // 0..31
            const int c2 = (tid & 7) * 16;      // 0..112
            const f16* bsrc = B + (size_t)(k0 + r2) * ldb + col0 + c2;
            f16 tmp[16];
            ((int4*)tmp)[0] = ((const int4*)bsrc)[0];
            ((int4*)tmp)[1] = ((const int4*)bsrc)[1];
            if (k0 + 32 < K) __builtin_prefetch(bsrc + 32 * ldb, 0, 0);
#pragma unroll
            for (int i = 0; i < 16; ++i) lds_bt[c2 + i][r2] = tmp[i];
        }
        wait_async0();
        __syncthreads();

        v16h af[2], bf[4];
#pragma unroll
        for (int ti = 0; ti < 2; ++ti) {
            const f16* p = &lds_a[wr + ti * 16 + lh][hi ? 8 : 0];
#pragma unroll
            for (int i = 0; i < 8; ++i) { af[ti][i] = p[i]; af[ti][i + 8] = p[i + 16]; }
        }
#pragma unroll
        for (int tj = 0; tj < 4; ++tj) {
            const f16* p = &lds_bt[wc + tj * 16 + lh][hi ? 8 : 0];
#pragma unroll
            for (int i = 0; i < 8; ++i) { bf[tj][i] = p[i]; bf[tj][i + 8] = p[i + 16]; }
        }
#pragma unroll
        for (int ti = 0; ti < 2; ++ti)
#pragma unroll
            for (int tj = 0; tj < 4; ++tj)
                acc[ti][tj] = wmma_f16(af[ti], bf[tj], acc[ti][tj]);
        __syncthreads();
    }

#pragma unroll
    for (int ti = 0; ti < 2; ++ti)
#pragma unroll
        for (int tj = 0; tj < 4; ++tj)
#pragma unroll
            for (int r = 0; r < 8; ++r) {
                const int gm = row0 + wr + ti * 16 + (hi ? r + 8 : r);
                const int gn = col0 + wc + tj * 16 + lh;
                C[(size_t)gm * ldc + gn] = (OutT)acc[ti][tj][r];
            }
}

// ---------------------------------------------------------------------------
// In-place RoPE on (B, S, H, 128) f16, first 64 dims rotated
// ---------------------------------------------------------------------------
__global__ void rope_kernel(f16* __restrict__ X) {
    size_t idx = (size_t)blockIdx.x * blockDim.x + threadIdx.x;
    const int    j   = (int)(idx & 31);
    const size_t th  = idx >> 5;            // (b*S + s)*H + h
    const size_t tok = th / HN;
    const int    pos = (int)(tok % SN);
    f16* base = X + th * DN;
    const float inv_freq = __powf(10000.0f, -(float)(2 * j) / (float)RDN);
    float c, sn;
    __sincosf((float)pos * inv_freq, &sn, &c);
    const float x0 = (float)base[j];
    const float x1 = (float)base[j + 32];
    base[j]      = (f16)(x0 * c - x1 * sn);
    base[j + 32] = (f16)(x1 * c + x0 * sn);
}

// ---------------------------------------------------------------------------
// Extract V: v[tok,h,d] = kv[tok, KR + h*2*D + d]
// ---------------------------------------------------------------------------
__global__ void extract_v(const f16* __restrict__ kv, f16* __restrict__ v) {
    size_t idx = (size_t)blockIdx.x * blockDim.x + threadIdx.x;
    if (idx >= (size_t)TOK * HN * DN) return;
    const int    d   = (int)(idx % DN);
    const size_t r   = idx / DN;
    const int    h   = (int)(r % HN);
    const size_t tok = r / HN;
    v[idx] = kv[tok * KVW + KRN + (size_t)h * 2 * DN + d];
}

// ---------------------------------------------------------------------------
// Flash attention, causal. grid = (S/64, H, B), 128 threads (4 waves),
// each wave owns 16 query rows. WMMA for QK^T and PV; async K staging.
// ---------------------------------------------------------------------------
__global__ __launch_bounds__(128) void attn_kernel(const f16* __restrict__ Q,
                                                   const f16* __restrict__ K,
                                                   const f16* __restrict__ V,
                                                   f16* __restrict__ Y) {
    const int qblk = blockIdx.x;
    const int h    = blockIdx.y;
    const int bb   = blockIdx.z;
    const int tid  = threadIdx.x;
    const int wave = tid >> 5, lane = tid & 31;
    const int lh   = lane & 15, hi = lane >> 4;
    const int q0   = qblk * 64;
    const int wq0  = q0 + wave * 16;

    __shared__ f16 lds_k[64][DN];       // [key][d]  == transposed-B for Q*K^T
    __shared__ f16 lds_vt[DN][64];      // [d][key]  == transposed-B for P*V
    __shared__ f16 lds_p[4][16][64];    // per-wave P staging

    // ---- preload Q fragments directly from global ----
    v16h qf[4];
    {
        const int row = wq0 + lh;
        const f16* qrow = Q + (((size_t)(bb * SN + row) * HN + h) * DN);
        const int kb = hi ? 8 : 0;
#pragma unroll
        for (int c = 0; c < 4; ++c) {
            const f16* p = qrow + c * 32 + kb;
#pragma unroll
            for (int i = 0; i < 8; ++i) { qf[c][i] = p[i]; qf[c][i + 8] = p[i + 16]; }
        }
    }

    v8f o_acc[8];
    {
        v8f z = {};
#pragma unroll
        for (int t = 0; t < 8; ++t) o_acc[t] = z;
    }
    float m_i[8], l_i[8];
#pragma unroll
    for (int r = 0; r < 8; ++r) { m_i[r] = -1e30f; l_i[r] = 0.f; }

    const float scale = 0.08838834764831845f; // 1/sqrt(128)
    const int nkb = qblk + 1;

    for (int j = 0; j < nkb; ++j) {
        const int k0 = j * 64;
        {   // K tile via async global->LDS (layout preserved)
            const int key = tid >> 1;
            const int d0  = (tid & 1) * 64;
            const f16* ks = K + (((size_t)(bb * SN + k0 + key) * HN + h) * DN) + d0;
            unsigned lk = lds_off_u32(&lds_k[key][d0]);
#pragma unroll
            for (int i = 0; i < 8; ++i)
                async_b128(lk + (unsigned)(i * 16), ks + i * 8);
            // V tile transposed (manual)
            const f16* vs = V + (((size_t)(bb * SN + k0 + key) * HN + h) * DN) + d0;
#pragma unroll
            for (int i = 0; i < 64; ++i) lds_vt[d0 + i][key] = vs[i];
        }
        wait_async0();
        __syncthreads();

        // ---- S = Q * K^T ----
        v8f sacc[4];
        { v8f z = {}; sacc[0] = z; sacc[1] = z; sacc[2] = z; sacc[3] = z; }
#pragma unroll
        for (int c = 0; c < 4; ++c) {
#pragma unroll
            for (int tj = 0; tj < 4; ++tj) {
                v16h bf;
                const f16* p = &lds_k[tj * 16 + lh][c * 32 + (hi ? 8 : 0)];
#pragma unroll
                for (int i = 0; i < 8; ++i) { bf[i] = p[i]; bf[i + 8] = p[i + 16]; }
                sacc[tj] = wmma_f16(qf[c], bf, sacc[tj]);
            }
        }

        // ---- scale + causal mask + row max ----
        float rowmax[8];
#pragma unroll
        for (int r = 0; r < 8; ++r) rowmax[r] = -1e30f;
#pragma unroll
        for (int tj = 0; tj < 4; ++tj)
#pragma unroll
            for (int r = 0; r < 8; ++r) {
                const int qi = wq0 + (hi ? r + 8 : r);
                const int ki = k0 + tj * 16 + lh;
                float sv = sacc[tj][r] * scale;
                if (ki > qi) sv = -1e30f;
                sacc[tj][r] = sv;
                rowmax[r] = fmaxf(rowmax[r], sv);
            }
#pragma unroll
        for (int r = 0; r < 8; ++r) {
            rowmax[r] = fmaxf(rowmax[r], __shfl_xor(rowmax[r], 1, 16));
            rowmax[r] = fmaxf(rowmax[r], __shfl_xor(rowmax[r], 2, 16));
            rowmax[r] = fmaxf(rowmax[r], __shfl_xor(rowmax[r], 4, 16));
            rowmax[r] = fmaxf(rowmax[r], __shfl_xor(rowmax[r], 8, 16));
        }

        // ---- online softmax update ----
        float alpha[8], rowsum[8];
#pragma unroll
        for (int r = 0; r < 8; ++r) {
            const float nm = fmaxf(m_i[r], rowmax[r]);
            alpha[r] = __expf(m_i[r] - nm);
            m_i[r] = nm;
            rowsum[r] = 0.f;
        }
#pragma unroll
        for (int tj = 0; tj < 4; ++tj)
#pragma unroll
            for (int r = 0; r < 8; ++r) {
                const float pv = __expf(sacc[tj][r] - m_i[r]);
                rowsum[r] += pv;
                lds_p[wave][hi ? r + 8 : r][tj * 16 + lh] = (f16)pv;
            }
#pragma unroll
        for (int r = 0; r < 8; ++r) {
            rowsum[r] += __shfl_xor(rowsum[r], 1, 16);
            rowsum[r] += __shfl_xor(rowsum[r], 2, 16);
            rowsum[r] += __shfl_xor(rowsum[r], 4, 16);
            rowsum[r] += __shfl_xor(rowsum[r], 8, 16);
            l_i[r] = l_i[r] * alpha[r] + rowsum[r];
        }
#pragma unroll
        for (int t = 0; t < 8; ++t)
#pragma unroll
            for (int r = 0; r < 8; ++r) o_acc[t][r] *= alpha[r];
        __syncthreads();   // P visible, lds_k reads done

        // ---- O += P * V ----
#pragma unroll
        for (int c2 = 0; c2 < 2; ++c2) {
            v16h pf;
            {
                const f16* p = &lds_p[wave][lh][c2 * 32 + (hi ? 8 : 0)];
#pragma unroll
                for (int i = 0; i < 8; ++i) { pf[i] = p[i]; pf[i + 8] = p[i + 16]; }
            }
#pragma unroll
            for (int t = 0; t < 8; ++t) {
                v16h vf;
                const f16* p = &lds_vt[t * 16 + lh][c2 * 32 + (hi ? 8 : 0)];
#pragma unroll
                for (int i = 0; i < 8; ++i) { vf[i] = p[i]; vf[i + 8] = p[i + 16]; }
                o_acc[t] = wmma_f16(pf, vf, o_acc[t]);
            }
        }
        __syncthreads();   // before next iteration overwrites lds_k / lds_vt
    }

    // ---- epilogue: normalize and store y (f16) ----
#pragma unroll
    for (int t = 0; t < 8; ++t)
#pragma unroll
        for (int r = 0; r < 8; ++r) {
            const int qi = wq0 + (hi ? r + 8 : r);
            f16* dst = Y + (((size_t)(bb * SN + qi) * HN + h) * DN) + t * 16 + lh;
            *dst = (f16)(o_acc[t][r] / l_i[r]);
        }
}

// ---------------------------------------------------------------------------
// Host-side launcher
// ---------------------------------------------------------------------------
extern "C" void kernel_launch(void* const* d_in, const int* in_sizes, int n_in,
                              void* d_out, int out_size, void* d_ws, size_t ws_size,
                              hipStream_t stream) {
    const float* x     = (const float*)d_in[0];
    const float* g_q   = (const float*)d_in[1];
    const float* g_kv  = (const float*)d_in[2];
    const float* w_qa  = (const float*)d_in[3];
    const float* w_qb  = (const float*)d_in[4];
    const float* w_kva = (const float*)d_in[5];
    const float* w_kvb = (const float*)d_in[6];
    const float* w_o   = (const float*)d_in[7];
    float* out = (float*)d_out;

    char* base = (char*)d_ws;
    size_t off = 0;
    auto alloc = [&](size_t elems) -> f16* {
        f16* p = (f16*)(base + off);
        off = (off + elems * sizeof(f16) + 255) & ~(size_t)255;
        return p;
    };
    f16* ln_q   = alloc((size_t)TOK * EN);
    f16* ln_kv  = alloc((size_t)TOK * EN);
    f16* wqa16  = alloc((size_t)EN * QRN);
    f16* wqb16  = alloc((size_t)QRN * EN);
    f16* wkva16 = alloc((size_t)EN * KVW);
    f16* wkvb16 = alloc((size_t)KRN * EN);
    f16* wo16   = alloc((size_t)EN * EN);
    f16* qlat16 = alloc((size_t)TOK * QRN);
    f16* q16    = alloc((size_t)TOK * EN);
    f16* kv16   = alloc((size_t)TOK * KVW);
    f16* k16    = alloc((size_t)TOK * EN);
    f16* v16    = alloc((size_t)TOK * EN);
    f16* y16    = alloc((size_t)TOK * EN);

    auto cvt = [&](const float* s, f16* d, size_t n) {
        cvt_f32_f16<<<dim3((unsigned)((n + 255) / 256)), dim3(256), 0, stream>>>(s, d, n);
    };
    cvt(w_qa,  wqa16,  (size_t)EN * QRN);
    cvt(w_qb,  wqb16,  (size_t)QRN * EN);
    cvt(w_kva, wkva16, (size_t)EN * KVW);
    cvt(w_kvb, wkvb16, (size_t)KRN * EN);
    cvt(w_o,   wo16,   (size_t)EN * EN);

    // LayerNorm (both gains fused)
    ln_kernel<<<dim3(TOK), dim3(256), 0, stream>>>(x, g_q, g_kv, ln_q, ln_kv);

    // q_lat = ln_q @ w_qa   [4096 x 64]   (N=64 -> 64-wide kernel)
    gemm64_f16<f16><<<dim3(QRN / 64, TOK / 64), dim3(128), 0, stream>>>(
        ln_q, EN, wqa16, QRN, qlat16, QRN, TOK, QRN, EN);

    // q = q_lat @ w_qb      [4096 x 2048]
    gemm128_f16<f16><<<dim3(EN / 128, TOK / 128), dim3(256), 0, stream>>>(
        qlat16, QRN, wqb16, EN, q16, EN, TOK, EN, QRN);

    // kv = ln_kv @ w_kva    [4096 x 4224]
    gemm128_f16<f16><<<dim3(KVW / 128, TOK / 128), dim3(256), 0, stream>>>(
        ln_kv, EN, wkva16, KVW, kv16, KVW, TOK, KVW, EN);

    // k = kv[:, :128] @ w_kvb  [4096 x 2048]
    gemm128_f16<f16><<<dim3(EN / 128, TOK / 128), dim3(256), 0, stream>>>(
        kv16, KVW, wkvb16, EN, k16, EN, TOK, EN, KRN);

    // RoPE on q and k (in place)
    {
        const size_t nthr = (size_t)TOK * HN * 32;
        rope_kernel<<<dim3((unsigned)(nthr / 256)), dim3(256), 0, stream>>>(q16);
        rope_kernel<<<dim3((unsigned)(nthr / 256)), dim3(256), 0, stream>>>(k16);
    }

    // extract V
    {
        const size_t n = (size_t)TOK * HN * DN;
        extract_v<<<dim3((unsigned)((n + 255) / 256)), dim3(256), 0, stream>>>(kv16, v16);
    }

    // flash attention
    attn_kernel<<<dim3(SN / 64, HN, BN), dim3(128), 0, stream>>>(q16, k16, v16, y16);

    // out = y @ w_o  (fp32 output)
    gemm128_f16<float><<<dim3(EN / 128, TOK / 128), dim3(256), 0, stream>>>(
        y16, EN, wo16, EN, out, EN, TOK, EN, EN);
}
